// DecoderMultiheadAttention_27882927685923
// MI455X (gfx1250) — matrix operand order repaired and checked by reading, not compiled, not run on hardware
//
#include <hip/hip_runtime.h>
#include <hip/hip_bf16.h>

// ---------------------------------------------------------------------------
// DecoderMultiheadAttention for MI455X (gfx1250), wave32 + WMMA bf16.
// B=2, S=2048, D_MODEL=1024, N_HEAD=16, HEAD_DIM=64.
// d_out = [ out (B,S,D) fp32 | attn_weights (B,S,S,NH) fp32 ]
//
// - All matmuls: v_wmma_f32_16x16x32_bf16, fp32 accumulate.
// - GEMM: 16x64 tile per wave, manually double-buffered k-loop so next-step
//   loads are in flight during this step's WMMAs (staggered s_wait_loadcnt).
// - V produced TRANSPOSED (d_model major) so attention's attn@V
//   B-fragments are contiguous 32B/lane loads (global_load_b128).
// ---------------------------------------------------------------------------

typedef __attribute__((ext_vector_type(16))) __bf16 v16bf;
typedef __attribute__((ext_vector_type(8)))  float  v8f;

#define B_     2
#define SEQ    2048
#define DMODEL 1024
#define NHEAD  16
#define HDIM   64
#define MROWS  (B_ * SEQ)
#define SCALE_Q 0.125f                 // HEAD_DIM^-0.5
#define NEG_INF (-__builtin_inff())

// ---- WMMA fragment loaders (contiguous k = 16*hi + s, same in A and B, wave32) ---------

// A-matrix 16x32 bf16: lane L(&15) = row M; lane>>4 selects K-half.
static __device__ __forceinline__ v16bf load_a_frag(const __bf16* __restrict__ A,
                                                    int lda, int m0, int k0, int lane) {
    const int row = m0 + (lane & 15);
    const int hi  = lane >> 4;
    const __bf16* p0 = A + (size_t)row * lda + k0 + hi * 16;
    const __bf16* p1 = p0 + 8;
    v16bf a;
#pragma unroll
    for (int i = 0; i < 8; ++i) { a[i] = p0[i]; a[i + 8] = p1[i]; }
    return a;
}

// B-matrix 32x16 where B[k][n] = W[n][k] (W row-major NxK): lane holds
// column n = n0+(lane&15), K range [k0 + hi*16, +16) -> contiguous in W row.
static __device__ __forceinline__ v16bf load_bt_frag(const __bf16* __restrict__ W,
                                                     int ldw, int n0, int k0, int lane) {
    const int col = n0 + (lane & 15);
    const int hi  = lane >> 4;
    const __bf16* p = W + (size_t)col * ldw + k0 + hi * 16;
    v16bf b;
#pragma unroll
    for (int i = 0; i < 16; ++i) b[i] = p[i];
    return b;
}

static __device__ __forceinline__ v8f wmma_bf16(v16bf a, v16bf b, v8f c) {
    return __builtin_amdgcn_wmma_f32_16x16x32_bf16(false, a, false, b,
                                                   (short)0, c, false, false);
}

// ---- fp32 -> bf16 conversion ----------------------------------------------
__global__ void cvt_f32_to_bf16(const float* __restrict__ in,
                                __bf16* __restrict__ out, int n) {
    int i = blockIdx.x * 256 + threadIdx.x;
    if (i < n) out[i] = (__bf16)in[i];
}

// ---- GEMM:  out = (A @ W^T + bias) * scale --------------------------------
// A: MxK bf16 row-major; W: NxK bf16 row-major.
// Block = (32,8): 8 waves; each wave computes a 16x64 tile (4 accumulators,
// one shared A-frag per k-step). Block covers 16 x 512; grid (M/16, N/512).
// transpose_out: bf16 output stored column-major (outb[col*M + row]).
__global__ void gemm_bf16_wmma(const __bf16* __restrict__ A,
                               const __bf16* __restrict__ W,
                               const float* __restrict__ bias, float scale,
                               int M, int N, int K,
                               __bf16* __restrict__ outb,
                               float* __restrict__ outf,
                               int transpose_out) {
    const int lane = threadIdx.x;
    const int m0 = blockIdx.x * 16;
    const int n0 = blockIdx.y * 512 + threadIdx.y * 64;

    // warm the first tiles toward the WGP cache (no in-loop branches)
    __builtin_prefetch(A + (size_t)(m0 + (lane & 15)) * K, 0, 3);
    __builtin_prefetch(W + (size_t)(n0 + (lane & 15)) * K, 0, 3);
    __builtin_prefetch(W + (size_t)(n0 + 32 + (lane & 15)) * K, 0, 3);

    v8f acc0 = {}, acc1 = {}, acc2 = {}, acc3 = {};

    // prologue: fragments for k0 = 0
    v16bf a  = load_a_frag(A, K, m0, 0, lane);
    v16bf b0 = load_bt_frag(W, K, n0 +  0, 0, lane);
    v16bf b1 = load_bt_frag(W, K, n0 + 16, 0, lane);
    v16bf b2 = load_bt_frag(W, K, n0 + 32, 0, lane);
    v16bf b3 = load_bt_frag(W, K, n0 + 48, 0, lane);

    // pipelined steps: issue k0+32 loads, then WMMA on k0 fragments
    for (int k0 = 0; k0 < K - 32; k0 += 32) {
        v16bf an  = load_a_frag(A, K, m0, k0 + 32, lane);
        v16bf bn0 = load_bt_frag(W, K, n0 +  0, k0 + 32, lane);
        v16bf bn1 = load_bt_frag(W, K, n0 + 16, k0 + 32, lane);
        v16bf bn2 = load_bt_frag(W, K, n0 + 32, k0 + 32, lane);
        v16bf bn3 = load_bt_frag(W, K, n0 + 48, k0 + 32, lane);
        acc0 = wmma_bf16(a, b0, acc0);
        acc1 = wmma_bf16(a, b1, acc1);
        acc2 = wmma_bf16(a, b2, acc2);
        acc3 = wmma_bf16(a, b3, acc3);
        a = an; b0 = bn0; b1 = bn1; b2 = bn2; b3 = bn3;
    }
    // epilogue: last k-step
    acc0 = wmma_bf16(a, b0, acc0);
    acc1 = wmma_bf16(a, b1, acc1);
    acc2 = wmma_bf16(a, b2, acc2);
    acc3 = wmma_bf16(a, b3, acc3);

    const int hi = lane >> 4;
#pragma unroll
    for (int t = 0; t < 4; ++t) {
        const v8f acc = (t == 0) ? acc0 : (t == 1) ? acc1 : (t == 2) ? acc2 : acc3;
        const int col = n0 + t * 16 + (lane & 15);
        const float bv = bias[col];
#pragma unroll
        for (int r = 0; r < 8; ++r) {
            const int row = m0 + r + 8 * hi;        // C/D layout: VGPR r -> M
            const float v = (acc[r] + bv) * scale;
            if (outf) outf[(size_t)row * N + col] = v;
            if (outb) {
                if (transpose_out) outb[(size_t)col * M + row] = (__bf16)v;
                else               outb[(size_t)row * N + col] = (__bf16)v;
            }
        }
    }
}

// ---- Fused attention per (batch, head, 16-query tile) ---------------------
// 8 waves; wave w owns K slice [w*256, w*256+256). Full 16x2048 score row in
// LDS -> block softmax -> write weights (B,Q,K,NH) -> attn @ V via WMMA.
// vt: transposed V, layout (DMODEL, MROWS): vt[(h*64+d)*MROWS + b*SEQ + k].
__global__ void attn_fused(const __bf16* __restrict__ qb,
                           const __bf16* __restrict__ kb,
                           const __bf16* __restrict__ vt,
                           const unsigned char* __restrict__ mask,
                           float* __restrict__ attn_w,
                           __bf16* __restrict__ ao) {
    extern __shared__ float smem[];
    float* sc     = smem;                  // 16 * SEQ scores/weights
    float* outred = sc + 16 * SEQ;         // 16 * 64 output tile
    float* rowred = outred + 16 * HDIM;    // 16 * 16 reduction scratch
    float* rowmax = rowred + 256;          // 16
    float* rowinv = rowmax + 16;           // 16

    const int tid  = threadIdx.x;
    const int lane = tid & 31;
    const int wave = tid >> 5;
    const int hi   = lane >> 4;
    const int qt = blockIdx.x, h = blockIdx.y, b = blockIdx.z;

    const __bf16* qbase  = qb + ((size_t)(b * SEQ) + qt * 16) * DMODEL + h * HDIM;
    const __bf16* kbase  = kb + (size_t)(b * SEQ) * DMODEL + h * HDIM;
    const __bf16* vtbase = vt + (size_t)(h * HDIM) * MROWS + (size_t)b * SEQ;

    // warm this wave's first K-tile
    __builtin_prefetch(kbase + (size_t)(wave * 256 + (lane & 15)) * DMODEL, 0, 3);

    // q tile 16x64 (already scaled by 0.125 in projection)
    const v16bf a0 = load_a_frag(qbase, DMODEL, 0, 0, lane);
    const v16bf a1 = load_a_frag(qbase, DMODEL, 0, 32, lane);

    const unsigned char* mrow = mask + b * SEQ;

    // -- scores: 16 tiles of 16 keys per wave, 2 WMMAs each, pipelined -----
    {
        v16bf c0 = load_bt_frag(kbase, DMODEL, wave * 256, 0, lane);
        v16bf c1 = load_bt_frag(kbase, DMODEL, wave * 256, 32, lane);
        for (int kt = 0; kt < 15; ++kt) {
            const int kc0 = wave * 256 + kt * 16;
            v16bf n0f = load_bt_frag(kbase, DMODEL, kc0 + 16, 0, lane);
            v16bf n1f = load_bt_frag(kbase, DMODEL, kc0 + 16, 32, lane);
            v8f s = {};
            s = wmma_bf16(a0, c0, s);
            s = wmma_bf16(a1, c1, s);
            const int col = kc0 + (lane & 15);
            const bool msk = mrow[col] != 0;         // True -> -inf (reference)
#pragma unroll
            for (int r = 0; r < 8; ++r)
                sc[(r + 8 * hi) * SEQ + col] = msk ? NEG_INF : s[r];
            c0 = n0f; c1 = n1f;
        }
        // epilogue tile
        v8f s = {};
        s = wmma_bf16(a0, c0, s);
        s = wmma_bf16(a1, c1, s);
        const int col = wave * 256 + 240 + (lane & 15);
        const bool msk = mrow[col] != 0;
#pragma unroll
        for (int r = 0; r < 8; ++r)
            sc[(r + 8 * hi) * SEQ + col] = msk ? NEG_INF : s[r];
    }
    __syncthreads();

    // -- block softmax over K (rows=16, cols=2048, 16 threads per row) -----
    const int rr = tid >> 4;
    const int cl = tid & 15;
    {
        float mx = NEG_INF;
        for (int c = cl; c < SEQ; c += 16) mx = fmaxf(mx, sc[rr * SEQ + c]);
        rowred[rr * 16 + cl] = mx;
    }
    __syncthreads();
    if (tid < 16) {
        float m = NEG_INF;
        for (int j = 0; j < 16; ++j) m = fmaxf(m, rowred[tid * 16 + j]);
        rowmax[tid] = m;
    }
    __syncthreads();
    {
        const float m = rowmax[rr];
        float sm = 0.f;
        for (int c = cl; c < SEQ; c += 16) sm += __expf(sc[rr * SEQ + c] - m);
        rowred[rr * 16 + cl] = sm;
    }
    __syncthreads();
    if (tid < 16) {
        float s = 0.f;
        for (int j = 0; j < 16; ++j) s += rowred[tid * 16 + j];
        rowinv[tid] = 1.f / s;
    }
    __syncthreads();
    {   // normalize in LDS + write attn_weights (B,Q,K,NH)
        const float m = rowmax[rr], inv = rowinv[rr];
        const int q = qt * 16 + rr;
        for (int c = cl; c < SEQ; c += 16) {
            const float w = __expf(sc[rr * SEQ + c] - m) * inv;
            sc[rr * SEQ + c] = w;
            attn_w[(((size_t)b * SEQ + q) * SEQ + c) * NHEAD + h] = w;
        }
    }
    __syncthreads();

    // -- out tile: weights(16 x 2048) @ v(2048 x 64); pipelined ------------
    v8f o0 = {}, o1 = {}, o2 = {}, o3 = {};
    {
        const int m = lane & 15;
        const int kbeg = wave * 256;
        v16bf vb0 = load_bt_frag(vtbase, MROWS,  0, kbeg, lane);
        v16bf vb1 = load_bt_frag(vtbase, MROWS, 16, kbeg, lane);
        v16bf vb2 = load_bt_frag(vtbase, MROWS, 32, kbeg, lane);
        v16bf vb3 = load_bt_frag(vtbase, MROWS, 48, kbeg, lane);
        for (int kc = kbeg; kc < kbeg + 256 - 32; kc += 32) {
            v16bf nb0 = load_bt_frag(vtbase, MROWS,  0, kc + 32, lane);
            v16bf nb1 = load_bt_frag(vtbase, MROWS, 16, kc + 32, lane);
            v16bf nb2 = load_bt_frag(vtbase, MROWS, 32, kc + 32, lane);
            v16bf nb3 = load_bt_frag(vtbase, MROWS, 48, kc + 32, lane);
            v16bf a;
#pragma unroll
            for (int i = 0; i < 8; ++i) {
                a[i]     = (__bf16)sc[m * SEQ + kc + hi * 16 + i];
                a[i + 8] = (__bf16)sc[m * SEQ + kc + hi * 16 + 8 + i];
            }
            o0 = wmma_bf16(a, vb0, o0);
            o1 = wmma_bf16(a, vb1, o1);
            o2 = wmma_bf16(a, vb2, o2);
            o3 = wmma_bf16(a, vb3, o3);
            vb0 = nb0; vb1 = nb1; vb2 = nb2; vb3 = nb3;
        }
        {   // epilogue k-step
            const int kc = kbeg + 256 - 32;
            v16bf a;
#pragma unroll
            for (int i = 0; i < 8; ++i) {
                a[i]     = (__bf16)sc[m * SEQ + kc + hi * 16 + i];
                a[i + 8] = (__bf16)sc[m * SEQ + kc + hi * 16 + 8 + i];
            }
            o0 = wmma_bf16(a, vb0, o0);
            o1 = wmma_bf16(a, vb1, o1);
            o2 = wmma_bf16(a, vb2, o2);
            o3 = wmma_bf16(a, vb3, o3);
        }
    }
    for (int i = tid; i < 16 * HDIM; i += 256) outred[i] = 0.f;
    __syncthreads();
    {
        const int c0 = lane & 15;
#pragma unroll
        for (int r = 0; r < 8; ++r) {
            const int row = r + 8 * hi;
            atomicAdd(&outred[row * HDIM +  0 + c0], o0[r]);
            atomicAdd(&outred[row * HDIM + 16 + c0], o1[r]);
            atomicAdd(&outred[row * HDIM + 32 + c0], o2[r]);
            atomicAdd(&outred[row * HDIM + 48 + c0], o3[r]);
        }
    }
    __syncthreads();
    for (int i = tid; i < 16 * HDIM; i += 256) {
        const int row = i >> 6, c = i & 63;
        ao[((size_t)(b * SEQ) + qt * 16 + row) * DMODEL + h * HDIM + c] =
            (__bf16)outred[i];
    }
}

// ---------------------------------------------------------------------------
extern "C" void kernel_launch(void* const* d_in, const int* in_sizes, int n_in,
                              void* d_out, int out_size, void* d_ws, size_t ws_size,
                              hipStream_t stream) {
    (void)in_sizes; (void)n_in; (void)out_size; (void)ws_size;

    const float* query = (const float*)d_in[0];
    const float* key   = (const float*)d_in[1];
    const float* value = (const float*)d_in[2];
    const unsigned char* mask = (const unsigned char*)d_in[3];
    const float* Wq = (const float*)d_in[4];
    const float* bq = (const float*)d_in[5];
    const float* Wk = (const float*)d_in[6];
    const float* bk = (const float*)d_in[7];
    const float* Wv = (const float*)d_in[8];
    const float* bv = (const float*)d_in[9];
    const float* Wo = (const float*)d_in[10];
    const float* bo = (const float*)d_in[11];

    float* out_f  = (float*)d_out;
    float* attn_w = out_f + (size_t)B_ * SEQ * DMODEL;

    const size_t NX = (size_t)MROWS * DMODEL;     // 4,194,304
    const size_t NW = (size_t)DMODEL * DMODEL;    // 1,048,576
    __bf16* ws = (__bf16*)d_ws;
    __bf16* xq = ws; ws += NX;
    __bf16* xk = ws; ws += NX;
    __bf16* xv = ws; ws += NX;
    __bf16* wq = ws; ws += NW;
    __bf16* wk = ws; ws += NW;
    __bf16* wv = ws; ws += NW;
    __bf16* wo = ws; ws += NW;
    __bf16* qb = ws; ws += NX;
    __bf16* kb = ws; ws += NX;
    __bf16* vt = ws; ws += NX;                    // transposed V (DMODEL, MROWS)
    __bf16* ao = ws; ws += NX;

    // 1) fp32 -> bf16 conversions
    cvt_f32_to_bf16<<<(int)(NX / 256), 256, 0, stream>>>(query, xq, (int)NX);
    cvt_f32_to_bf16<<<(int)(NX / 256), 256, 0, stream>>>(key,   xk, (int)NX);
    cvt_f32_to_bf16<<<(int)(NX / 256), 256, 0, stream>>>(value, xv, (int)NX);
    cvt_f32_to_bf16<<<(int)(NW / 256), 256, 0, stream>>>(Wq, wq, (int)NW);
    cvt_f32_to_bf16<<<(int)(NW / 256), 256, 0, stream>>>(Wk, wk, (int)NW);
    cvt_f32_to_bf16<<<(int)(NW / 256), 256, 0, stream>>>(Wv, wv, (int)NW);
    cvt_f32_to_bf16<<<(int)(NW / 256), 256, 0, stream>>>(Wo, wo, (int)NW);

    // 2) Q/K/V projections (q gets the 1/sqrt(HEAD_DIM) scale baked in;
    //    V stored transposed for contiguous attention B-fragments)
    dim3 gblk(32, 8);
    dim3 ggrid(MROWS / 16, DMODEL / 512);
    gemm_bf16_wmma<<<ggrid, gblk, 0, stream>>>(xq, wq, bq, SCALE_Q,
                                               MROWS, DMODEL, DMODEL, qb, nullptr, 0);
    gemm_bf16_wmma<<<ggrid, gblk, 0, stream>>>(xk, wk, bk, 1.0f,
                                               MROWS, DMODEL, DMODEL, kb, nullptr, 0);
    gemm_bf16_wmma<<<ggrid, gblk, 0, stream>>>(xv, wv, bv, 1.0f,
                                               MROWS, DMODEL, DMODEL, vt, nullptr, 1);

    // 3) fused attention: grid (q-tiles, heads, batch)
    dim3 agrid(SEQ / 16, NHEAD, B_);
    const size_t shbytes = (size_t)(16 * SEQ + 16 * HDIM + 256 + 16 + 16) * sizeof(float);
    attn_fused<<<agrid, 256, shbytes, stream>>>(qb, kb, vt, mask, attn_w, ao);

    // 4) output projection -> fp32 out
    gemm_bf16_wmma<<<ggrid, gblk, 0, stream>>>(ao, wo, bo, 1.0f,
                                               MROWS, DMODEL, DMODEL, nullptr, out_f, 0);
}
